// AttentionHead_52759378264083
// MI455X (gfx1250) — matrix operand order repaired and checked by reading, not compiled, not run on hardware
//
#include <hip/hip_runtime.h>
#include <hip/hip_bf16.h>

// ---------------------------------------------------------------------------
// AttentionHead for MI455X (gfx1250, wave32, WMMA bf16 16x16x32, f32 accum)
//   q,k,v = x@W^T ; scores = (q k^T)/sqrt(D) + causal ; attn = softmax
//   out = LayerNorm(attn@v + q)
// S=8192, D=512.  Flash-attention structure; Q/K tiles staged to LDS with the
// Tensor Data Mover (tensor_load_to_lds + s_wait_tensorcnt).
// ---------------------------------------------------------------------------

typedef __attribute__((ext_vector_type(16))) __bf16          v16bf;
typedef __attribute__((ext_vector_type(16))) unsigned short  v16u;
typedef __attribute__((ext_vector_type(8)))  float           v8f;
typedef __attribute__((ext_vector_type(4)))  unsigned int    v4ui;
typedef __attribute__((ext_vector_type(8)))  int             v8si;
typedef __attribute__((ext_vector_type(4)))  int             v4si;
typedef unsigned short u16;

#define S_CTX  8192
#define D_DIM  512
#define QK_SCALE 0.044194173824159216f   // 1/sqrt(512)

#if defined(__AMDGCN__) && __has_builtin(__builtin_amdgcn_tensor_load_to_lds)
#define USE_TDM 1
#else
#define USE_TDM 0
#endif

// ---- helpers --------------------------------------------------------------
__device__ __forceinline__ u16 f2bf(float f) {
  unsigned int u = __float_as_uint(f);
  u += 0x7FFFu + ((u >> 16) & 1u);       // round-to-nearest-even
  return (u16)(u >> 16);
}

// A-matrix fragment (16x32 bf16, M x K), row-major source, ld in elements.
// ISA layout: lane<16 : row=lane,    elems 0..7 = K0..7,  8..15 = K16..23
//             lane>=16: row=lane-16, elems 0..7 = K8..15, 8..15 = K24..31
__device__ __forceinline__ v16bf frag_a(const u16* base, int ld, int lane) {
  int m  = lane & 15;
  int kh = (lane >> 4) & 1;
  const u16* p = base + m * ld + kh * 8;
  v16u r;
#pragma unroll
  for (int i = 0; i < 8; ++i) { r[i] = p[i]; r[i + 8] = p[i + 16]; }
  return __builtin_bit_cast(v16bf, r);
}

// B-matrix fragment (K=32 x N=16 bf16).  Source is the [N x K] (i.e. B^T)
// matrix row-major.  ISA layout: lane<16 : col=lane, elems 0..7 = K0..7, 8..15 = K16..23;
// lane>=16: col=lane-16, elems 0..7 = K8..15, 8..15 = K24..31.
__device__ __forceinline__ v16bf frag_b(const u16* base, int ld, int lane) {
  int n  = lane & 15;
  int kh = (lane >> 4) & 1;
  const u16* p = base + n * ld + kh * 8;
  v16u r;
#pragma unroll
  for (int i = 0; i < 8; ++i) { r[i] = p[i]; r[i + 8] = p[i + 16]; }
  return __builtin_bit_cast(v16bf, r);
}

__device__ __forceinline__ v8f wmma_bf16(v16bf a, v16bf b, v8f c) {
  return __builtin_amdgcn_wmma_f32_16x16x32_bf16(
      false, a, false, b, (short)0, c, false, false);
}

#if USE_TDM
// TDM: load a 2D bf16 tile [tile_rows x 512] (row stride 512 elems) from
// global memory into LDS at lds_byte_off, padding each 1024B row with 16B
// (pad_interval code 7 = 256 DWORDs, pad_amount code 3 = 4 DWORDs) so the
// LDS row stride is 520 ushorts.  Issued once per wave (EXEC ignored).
__device__ __forceinline__ void tdm_load_rows_bf16(const u16* gptr,
                                                   unsigned lds_byte_off,
                                                   int tile_rows) {
  unsigned long long ga = (unsigned long long)gptr;
  v4ui g0;
  g0[0] = 1u;                                        // count=1, user mode
  g0[1] = lds_byte_off;                              // lds_addr (bytes)
  g0[2] = (unsigned)(ga & 0xFFFFFFFFull);            // global_addr[31:0]
  g0[3] = (unsigned)((ga >> 32) & 0x1FFFFFFull)      // global_addr[56:32]
          | (2u << 30);                              // type = 2 ("image")
  v8si g1;
  g1[0] = (1 << 16)          // data_size = 1 -> 2 bytes
        | (1 << 20)          // pad_enable
        | (7 << 22)          // pad_interval code 7 -> 256 DWORDs (1024B row)
        | (3 << 25);         // pad_amount  code 3 -> 4 DWORDs (16B)
  g1[1] = (D_DIM & 0xFFFF) << 16;                    // tensor_dim0 = 512
  g1[2] = (S_CTX & 0xFFFF) << 16;                    // tensor_dim1 lo16
  g1[3] = ((S_CTX >> 16) & 0xFFFF) | (D_DIM << 16);  // dim1 hi | tile_dim0=512
  g1[4] = tile_rows & 0xFFFF;                        // tile_dim1
  g1[5] = D_DIM;                                     // tensor_dim0_stride lo32
  g1[6] = 0;
  g1[7] = 0;
  v4si z4; z4[0] = 0; z4[1] = 0; z4[2] = 0; z4[3] = 0;
  v8si z8;
#pragma unroll
  for (int i = 0; i < 8; ++i) z8[i] = 0;
  // clang-23 (therock-10.0) 6-arg form: (g0, g1, g2, g3, extra, cpol)
  __builtin_amdgcn_tensor_load_to_lds(g0, g1, z4, z4, z8, 0);
}
#endif

// ---------------------------------------------------------------------------
// Kernel 1: fused QKV projection.  y = x @ W^T  (W row-major [out][in])
// grid (64, 8, 3): z selects W / destination. block = 256 (8 waves).
// Tile: BM=128 rows (16/wave), BN=64 cols, BK=64.
// ---------------------------------------------------------------------------
#define LDX 72   // 64 + 8 pad (ushorts) -> 16B aligned rows, bank-friendly

__global__ __launch_bounds__(256) void qkv_kernel(
    const float* __restrict__ x,
    const float* __restrict__ Wq, const float* __restrict__ Wk,
    const float* __restrict__ Wv,
    u16* __restrict__ qb, u16* __restrict__ kb, u16* __restrict__ vb,
    float* __restrict__ qf)
{
  __shared__ u16 xs[128 * LDX];
  __shared__ u16 wsm[64 * LDX];

  const int tid  = threadIdx.x;
  const int wid  = tid >> 5;
  const int lane = tid & 31;
  const int m0   = blockIdx.x * 128;
  const int n0   = blockIdx.y * 64;
  const int z    = blockIdx.z;
  const float* W = (z == 0) ? Wq : (z == 1) ? Wk : Wv;

  v8f acc[4];
#pragma unroll
  for (int t = 0; t < 4; ++t)
#pragma unroll
    for (int r = 0; r < 8; ++r) acc[t][r] = 0.0f;

  for (int kb0 = 0; kb0 < D_DIM; kb0 += 64) {
    __syncthreads();
    // stage x block 128x64 (f32 -> bf16): 2048 float4 chunks
#pragma unroll
    for (int i = 0; i < 8; ++i) {
      int ch = tid + i * 256;
      int r  = ch >> 4;
      int c4 = (ch & 15) * 4;
      float4 f = *reinterpret_cast<const float4*>(x + (size_t)(m0 + r) * D_DIM + kb0 + c4);
      u16* d = &xs[r * LDX + c4];
      d[0] = f2bf(f.x); d[1] = f2bf(f.y); d[2] = f2bf(f.z); d[3] = f2bf(f.w);
    }
    // stage W block 64x64
#pragma unroll
    for (int i = 0; i < 4; ++i) {
      int ch = tid + i * 256;
      int r  = ch >> 4;
      int c4 = (ch & 15) * 4;
      float4 f = *reinterpret_cast<const float4*>(W + (size_t)(n0 + r) * D_DIM + kb0 + c4);
      u16* d = &wsm[r * LDX + c4];
      d[0] = f2bf(f.x); d[1] = f2bf(f.y); d[2] = f2bf(f.z); d[3] = f2bf(f.w);
    }
    __syncthreads();
#pragma unroll
    for (int nt = 0; nt < 4; ++nt) {
#pragma unroll
      for (int ks = 0; ks < 2; ++ks) {
        v16bf a = frag_a(&xs[(wid * 16) * LDX + ks * 32], LDX, lane);
        v16bf b = frag_b(&wsm[(nt * 16) * LDX + ks * 32], LDX, lane);
        acc[nt] = wmma_bf16(a, b, acc[nt]);
      }
    }
  }

  // write results.  C layout: lane<16 -> N=lane, M=r; lane>=16 -> M=r+8.
  const int hi8 = ((lane >> 4) & 1) * 8;
  const int nn  = lane & 15;
#pragma unroll
  for (int nt = 0; nt < 4; ++nt) {
#pragma unroll
    for (int r = 0; r < 8; ++r) {
      int row = m0 + wid * 16 + r + hi8;
      int col = n0 + nt * 16 + nn;
      float v = acc[nt][r];
      size_t idx = (size_t)row * D_DIM + col;
      if (z == 0) { qf[idx] = v; qb[idx] = f2bf(v * QK_SCALE); }
      else if (z == 1) kb[idx] = f2bf(v);
      else             vb[idx] = f2bf(v);
    }
  }
}

// ---------------------------------------------------------------------------
// Kernel 2: flash attention (causal) + residual(q) + LayerNorm.
// 256 workgroups x 256 threads. Each WG: 32 query rows (2 M-tiles).
// Q/K tiles DMA'd to LDS by the TDM; V staged transposed by VALU.
// Per 64-key block: 8 waves each compute one 16x16 score tile (K=512),
// cooperative online softmax, then each wave does P@V for its 64-col D slice.
// ---------------------------------------------------------------------------
#define BM 32
#define BN 64
#define LDQ 520   // 512 + 8 (ushorts)
#define LDK 520
#define LDV 72    // 64 + 8 (ushorts), V stored transposed [d][n]
#define LDSC 68   // 64 + 4 (floats)
#define LDP 72    // 64 + 8 (ushorts)
#define LDO 520   // floats

#define OFF_QS   0
#define OFF_KS   33280                    // 32*520*2
#define OFF_VT   (OFF_KS + 66560)         // + 64*520*2
#define OFF_SC   (OFF_VT + 73728)         // + 512*72*2
#define OFF_PS   (OFF_SC + 8704)          // + 32*68*4
#define OFF_ST   (OFF_PS + 4608)          // + 32*72*2
#define SMEM_BYTES (OFF_ST + 640 + 1024 + 1024)   // stats + red + red2
#define OFF_OUTB OFF_KS                   // out buffer overlays dead K tile

__global__ __launch_bounds__(256) void attn_kernel(
    const u16* __restrict__ qb, const u16* __restrict__ kb,
    const u16* __restrict__ vb, const float* __restrict__ qf,
    const float* __restrict__ gamma, const float* __restrict__ beta,
    float* __restrict__ out)
{
  extern __shared__ char smem[];
  u16*   qs   = reinterpret_cast<u16*>(smem + OFF_QS);
  u16*   ksld = reinterpret_cast<u16*>(smem + OFF_KS);
  u16*   vts  = reinterpret_cast<u16*>(smem + OFF_VT);
  float* sc   = reinterpret_cast<float*>(smem + OFF_SC);
  u16*   ps   = reinterpret_cast<u16*>(smem + OFF_PS);
  float* mrun = reinterpret_cast<float*>(smem + OFF_ST);
  float* lrun = mrun + 32;
  float* alph = lrun + 32;
  float* meanl= alph + 32;
  float* rstdl= meanl + 32;
  float* red  = rstdl + 32;     // 256 floats
  float* red2 = red + 256;      // 256 floats
  float* outb = reinterpret_cast<float*>(smem + OFF_OUTB);

  const int tid  = threadIdx.x;
  const int wid  = tid >> 5;
  const int lane = tid & 31;
  const int m0   = blockIdx.x * BM;
  const int hi8  = ((lane >> 4) & 1) * 8;
  const int nn   = lane & 15;

  if (tid < 32) { mrun[tid] = -1e30f; lrun[tid] = 0.0f; }

  // stage Q tile 32x512 (already bf16, pre-scaled by 1/sqrt(D))
#if USE_TDM
  if (wid == 0) {
    tdm_load_rows_bf16(qb + (size_t)m0 * D_DIM, OFF_QS, BM);
    __builtin_amdgcn_s_wait_tensorcnt(0);
  }
#else
#pragma unroll
  for (int i = 0; i < 8; ++i) {
    int ch = tid + i * 256;
    int r  = ch >> 6;
    int c8 = (ch & 63) * 8;
    *reinterpret_cast<uint4*>(&qs[r * LDQ + c8]) =
        *reinterpret_cast<const uint4*>(&qb[(size_t)(m0 + r) * D_DIM + c8]);
  }
#endif

  v8f o[8];
#pragma unroll
  for (int t = 0; t < 8; ++t)
#pragma unroll
    for (int r = 0; r < 8; ++r) o[t][r] = 0.0f;

  const int nblocks = (m0 + BM - 1) / BN + 1;

  for (int b = 0; b < nblocks; ++b) {
    const int n0 = b * BN;
    __syncthreads();

    // stage K block 64x512 row-major (TDM DMA)
#if USE_TDM
    if (wid == 0) tdm_load_rows_bf16(kb + (size_t)n0 * D_DIM, OFF_KS, BN);
#else
#pragma unroll
    for (int i = 0; i < 16; ++i) {
      int ch = tid + i * 256;
      int r  = ch >> 6;
      int c8 = (ch & 63) * 8;
      *reinterpret_cast<uint4*>(&ksld[r * LDK + c8]) =
          *reinterpret_cast<const uint4*>(&kb[(size_t)(n0 + r) * D_DIM + c8]);
    }
#endif
    // stage V block transposed: vts[d][n] = v[n0+n][d]
#pragma unroll
    for (int i = 0; i < 16; ++i) {
      int ch = tid + i * 256;
      int n  = ch >> 6;
      int d8 = (ch & 63) * 8;
      uint4 raw = *reinterpret_cast<const uint4*>(&vb[(size_t)(n0 + n) * D_DIM + d8]);
      const u16* rs = reinterpret_cast<const u16*>(&raw);
#pragma unroll
      for (int j = 0; j < 8; ++j) vts[(d8 + j) * LDV + n] = rs[j];
    }
#if USE_TDM
    if (wid == 0) __builtin_amdgcn_s_wait_tensorcnt(0);
#endif
    __syncthreads();

    // ---- scores: wave (mt,nt) computes S[16x16] over K=512 ----
    {
      const int mt = wid >> 2, nt = wid & 3;
      v8f c;
#pragma unroll
      for (int r = 0; r < 8; ++r) c[r] = 0.0f;
#pragma unroll
      for (int ksx = 0; ksx < 16; ++ksx) {
        v16bf a  = frag_a(&qs[(mt * 16) * LDQ + ksx * 32], LDQ, lane);
        v16bf bf = frag_b(&ksld[(nt * 16) * LDK + ksx * 32], LDK, lane);
        c = wmma_bf16(a, bf, c);
      }
#pragma unroll
      for (int r = 0; r < 8; ++r)
        sc[(mt * 16 + r + hi8) * LDSC + nt * 16 + nn] = c[r];
    }
    __syncthreads();

    // ---- online softmax (rows in LDS, 8 threads per row) ----
    {
      const int row = tid >> 3, seg = tid & 7;
      const int grow = m0 + row;
      float sv[8];
      float lm = -1e30f;
#pragma unroll
      for (int j = 0; j < 8; ++j) {
        float s = sc[row * LDSC + seg * 8 + j];
        if (n0 + seg * 8 + j > grow) s = -1e30f;   // strict causal mask
        sv[j] = s;
        lm = fmaxf(lm, s);
      }
      red[tid] = lm;
      __syncthreads();
      if (seg == 0) {
        float bm = red[row * 8];
#pragma unroll
        for (int j = 1; j < 8; ++j) bm = fmaxf(bm, red[row * 8 + j]);
        float mo = mrun[row];
        float mn = fmaxf(mo, bm);
        mrun[row] = mn;
        alph[row] = __expf(mo - mn);
      }
      __syncthreads();
      float mn = mrun[row];
      float lsum = 0.0f;
#pragma unroll
      for (int j = 0; j < 8; ++j) {
        float p = __expf(sv[j] - mn);
        lsum += p;
        ps[row * LDP + seg * 8 + j] = f2bf(p);
      }
      red[tid] = lsum;
      __syncthreads();
      if (seg == 0) {
        float bs = 0.0f;
#pragma unroll
        for (int j = 0; j < 8; ++j) bs += red[row * 8 + j];
        lrun[row] = lrun[row] * alph[row] + bs;
      }
      __syncthreads();
    }

    // ---- rescale O by alpha, accumulate P@V for this wave's D slice ----
    {
      const int d0 = wid * 64;
#pragma unroll
      for (int mt = 0; mt < 2; ++mt) {
#pragma unroll
        for (int r = 0; r < 8; ++r) {
          float a0 = alph[mt * 16 + r + hi8];
#pragma unroll
          for (int dt = 0; dt < 4; ++dt) o[mt * 4 + dt][r] *= a0;
        }
        // A (probability) fragments: load once, reuse across 4 D tiles
        v16bf a0 = frag_a(&ps[(mt * 16) * LDP +  0], LDP, lane);
        v16bf a1 = frag_a(&ps[(mt * 16) * LDP + 32], LDP, lane);
#pragma unroll
        for (int dt = 0; dt < 4; ++dt) {
          v16bf b0 = frag_b(&vts[(d0 + dt * 16) * LDV +  0], LDV, lane);
          v16bf b1 = frag_b(&vts[(d0 + dt * 16) * LDV + 32], LDV, lane);
          o[mt * 4 + dt] = wmma_bf16(a0, b0, o[mt * 4 + dt]);
          o[mt * 4 + dt] = wmma_bf16(a1, b1, o[mt * 4 + dt]);
        }
      }
    }
  }
  __syncthreads();

  // ---- O/l + residual(q) -> outb (overlays dead K tile) ----
  {
    const int d0 = wid * 64;
#pragma unroll
    for (int mt = 0; mt < 2; ++mt)
#pragma unroll
      for (int dt = 0; dt < 4; ++dt)
#pragma unroll
        for (int r = 0; r < 8; ++r) {
          int row = mt * 16 + r + hi8;
          int col = d0 + dt * 16 + nn;
          outb[row * LDO + col] =
              o[mt * 4 + dt][r] / lrun[row] + qf[(size_t)(m0 + row) * D_DIM + col];
        }
  }
  __syncthreads();

  // ---- LayerNorm stats ----
  {
    const int row = tid >> 3, seg = tid & 7;
    float s = 0.0f, ssq = 0.0f;
#pragma unroll 8
    for (int j = 0; j < 64; ++j) {
      float v = outb[row * LDO + seg * 64 + j];
      s += v; ssq += v * v;
    }
    red[tid] = s; red2[tid] = ssq;
    __syncthreads();
    if (seg == 0) {
      float ts = 0.0f, tss = 0.0f;
#pragma unroll
      for (int j = 0; j < 8; ++j) { ts += red[row * 8 + j]; tss += red2[row * 8 + j]; }
      float mean = ts * (1.0f / 512.0f);
      float var  = tss * (1.0f / 512.0f) - mean * mean;
      meanl[row] = mean;
      rstdl[row] = rsqrtf(var + 1e-5f);
    }
    __syncthreads();
  }

  // ---- normalize + write out (coalesced) ----
  for (int i = 0; i < 64; ++i) {
    int e = tid + i * 256;                 // 32*512 elements
    int row = e >> 9;
    int col = e & 511;
    float v = outb[row * LDO + col];
    out[(size_t)(m0 + row) * D_DIM + col] =
        (v - meanl[row]) * rstdl[row] * gamma[col] + beta[col];
  }
}

// ---------------------------------------------------------------------------
extern "C" void kernel_launch(void* const* d_in, const int* in_sizes, int n_in,
                              void* d_out, int out_size, void* d_ws, size_t ws_size,
                              hipStream_t stream) {
  (void)in_sizes; (void)n_in; (void)out_size; (void)ws_size;
  const float* x     = (const float*)d_in[0];
  // d_in[1] is the causal mask: deterministic strict upper-triangular, applied analytically.
  const float* Wq    = (const float*)d_in[2];
  const float* Wk    = (const float*)d_in[3];
  const float* Wv    = (const float*)d_in[4];
  const float* gamma = (const float*)d_in[5];
  const float* beta  = (const float*)d_in[6];
  float* out = (float*)d_out;

  // workspace: q_bf16 (8MB, pre-scaled) | k_bf16 (8MB) | v_bf16 (8MB) | q_f32 (16MB)
  char* ws = (char*)d_ws;
  u16*   qb = (u16*)(ws);
  u16*   kb = (u16*)(ws + (size_t)8  * 1024 * 1024);
  u16*   vb = (u16*)(ws + (size_t)16 * 1024 * 1024);
  float* qf = (float*)(ws + (size_t)24 * 1024 * 1024);

  dim3 g1(S_CTX / 128, D_DIM / 64, 3);
  qkv_kernel<<<g1, 256, 0, stream>>>(x, Wq, Wk, Wv, qb, kb, vb, qf);

  attn_kernel<<<dim3(S_CTX / BM), 256, SMEM_BYTES, stream>>>(
      qb, kb, vb, qf, gamma, beta, out);
}
